// HierarchicalVQ_46660524704245
// MI455X (gfx1250) — matrix operand order repaired and checked by reading, not compiled, not run on hardware
//
#include <hip/hip_runtime.h>
#include <hip/hip_bf16.h>
#include <float.h>

typedef __attribute__((ext_vector_type(16))) __bf16 v16bf;
typedef __attribute__((ext_vector_type(8)))  float  v8f;

#define NROWS 8192
#define DCOL  512
#define KCHUNKS 16   // 512 / 32

// -------- helpers --------

__device__ __forceinline__ unsigned long long packDI(float v, int i) {
    unsigned u = __float_as_uint(v);
    u = (u & 0x80000000u) ? ~u : (u | 0x80000000u);   // order-preserving f32 -> u32
    return ((unsigned long long)u << 32) | (unsigned)i;
}

__device__ __forceinline__ float sigmoid_f(float x) { return 1.0f / (1.0f + __expf(-x)); }

union APack { uint4 q[2]; v16bf v; };

// A-matrix 16x32 bf16 chunk per ISA layout:
//   lanes 0-15:  K = k0+0..7  and k0+16..23
//   lanes 16-31: K = k0+8..15 and k0+24..31
__device__ __forceinline__ v16bf load_a_chunk(const __bf16* rowp, int kc, int hx) {
    APack t;
    const __bf16* p = rowp + kc * 32 + hx * 8;
    t.q[0] = *reinterpret_cast<const uint4*>(p);
    t.q[1] = *reinterpret_cast<const uint4*>(p + 16);
    return t.v;
}

// B-matrix 32x16 bf16 chunk: lane holds col n = lane&15, same K order as A
//   K = k0 + hx*8 + 0..7 and k0 + 16 + hx*8 + 0..7
__device__ __forceinline__ v16bf load_b_chunk(const __bf16* colrow, int kc, int hx) {
    APack t;
    const __bf16* p = colrow + kc * 32 + hx * 8;
    t.q[0] = *reinterpret_cast<const uint4*>(p);
    t.q[1] = *reinterpret_cast<const uint4*>(p + 16);
    return t.v;
}

__device__ __forceinline__ void wait_async0() {
#if __has_builtin(__builtin_amdgcn_s_wait_asynccnt)
    __builtin_amdgcn_s_wait_asynccnt(0);
#else
    asm volatile("s_wait_asynccnt 0x0" ::: "memory");
#endif
}

// async 16B copy global -> LDS (GVS mode: sgpr base + 32-bit vgpr offset)
__device__ __forceinline__ void async_g2l_b128(unsigned lds_addr, unsigned goff,
                                               unsigned long long gbase) {
    asm volatile("global_load_async_to_lds_b128 %0, %1, %2"
                 :: "v"(lds_addr), "v"(goff), "s"(gbase) : "memory");
}

// two independent 16x16x512 WMMA accumulation streams sharing one A panel.
// cr0/cr1 point at the K=512 rows for column n = (lane&15) of each 16-col tile.
__device__ __forceinline__ void gemm2(const v16bf (&A)[KCHUNKS],
                                      const __bf16* cr0, const __bf16* cr1, int hx,
                                      v8f& out0, v8f& out1) {
    v8f acc0 = {}, acc1 = {};
    v16bf p0 = load_b_chunk(cr0, 0, hx);
    v16bf p1 = load_b_chunk(cr1, 0, hx);
    #pragma unroll
    for (int kc = 0; kc < KCHUNKS; ++kc) {
        v16bf n0 = p0, n1 = p1;
        if (kc + 1 < KCHUNKS) {
            n0 = load_b_chunk(cr0, kc + 1, hx);
            n1 = load_b_chunk(cr1, kc + 1, hx);
        }
        acc0 = __builtin_amdgcn_wmma_f32_16x16x32_bf16(false, A[kc], false, p0,
                                                       (short)0, acc0, false, false);
        acc1 = __builtin_amdgcn_wmma_f32_16x16x32_bf16(false, A[kc], false, p1,
                                                       (short)0, acc1, false, false);
        p0 = n0; p1 = n1;
    }
    out0 = acc0; out1 = acc1;
}

// -------- init: argmin accumulators + loss --------

__global__ void k_init(unsigned long long* pc, unsigned long long* pf, float* loss) {
    int i = blockIdx.x * blockDim.x + threadIdx.x;
    if (i < NROWS) { pc[i] = ~0ull; pf[i] = ~0ull; }
    if (i < 2) loss[i] = 0.0f;
}

// -------- convert f32 -> bf16, optional row-norm (one wave per row) --------

__global__ __launch_bounds__(256) void k_cvt_norm(const float* __restrict__ src, int srcld,
                                                  __bf16* __restrict__ dst,
                                                  float* __restrict__ norms, int cols) {
    const int lane = threadIdx.x & 31;
    const int wave = threadIdx.x >> 5;
    const int row  = blockIdx.x * 8 + wave;
    float s = 0.0f;
    for (int c = lane; c < cols; c += 32) {
        float v = src[(size_t)row * srcld + c];
        dst[(size_t)row * cols + c] = (__bf16)v;
        s += v * v;
    }
    if (norms) {
        #pragma unroll
        for (int o = 1; o < 32; o <<= 1) s += __shfl_xor(s, o, 32);
        if (lane == 0) norms[row] = s;
    }
}

// -------- VQ argmin GEMM: scores = X·Eᵀ via bf16 WMMA, dist = ‖e‖² - 2s --------
// block = 256 thr = 8 waves; wave owns 16 rows; block shares double-buffered
// 32-code LDS tiles filled with global_load_async_to_lds_b128 (ASYNCcnt).
// Each wave runs two independent WMMA streams over the two 16-code halves.
// grid = (NROWS/128, nsplit); splits combine via packed atomicMin per row.

__global__ __launch_bounds__(256) void k_vq(const __bf16* __restrict__ X,
                                            const __bf16* __restrict__ E,
                                            const float* __restrict__ enorm,
                                            unsigned long long* __restrict__ best,
                                            int codesPerSplit) {
    __shared__ __align__(32) __bf16 sE[2][32 * DCOL];   // 2 x 32 KB code tiles
    const int lane = threadIdx.x & 31;
    const int wave = threadIdx.x >> 5;
    const int hx   = lane >> 4;
    const int nl   = lane & 15;
    const int mbase = (blockIdx.x * 8 + wave) * 16;
    const int code0 = blockIdx.y * codesPerSplit;

    const unsigned long long Ebase = (unsigned long long)E;
    const unsigned ldsbase0 = (unsigned)(uintptr_t)(void*)&sE[0][0];
    const unsigned ldsbase1 = (unsigned)(uintptr_t)(void*)&sE[1][0];

    // issue async fill of one 32-code tile (32 KB): 8 x 16 B per thread
    auto issue_tile = [&](int t, int buf) {
        const unsigned gbyte = (unsigned)(code0 + t * 32) * (DCOL * 2);
        const unsigned lb    = buf ? ldsbase1 : ldsbase0;
        #pragma unroll
        for (int i = 0; i < 8; ++i) {
            const unsigned o = threadIdx.x * 16 + i * 4096;
            async_g2l_b128(lb + o, gbyte + o, Ebase);
        }
    };

    // A panel: this wave's 16 rows, full K=512, held in registers (16 v16bf)
    v16bf A[KCHUNKS];
    {
        const __bf16* rowp = X + (size_t)(mbase + nl) * DCOL;
        #pragma unroll
        for (int kc = 0; kc < KCHUNKS; ++kc) A[kc] = load_a_chunk(rowp, kc, hx);
    }

    float bv[8]; int bi[8];
    #pragma unroll
    for (int r = 0; r < 8; ++r) { bv[r] = FLT_MAX; bi[r] = 0x7fffffff; }

    const int ntiles = codesPerSplit >> 5;
    issue_tile(0, 0);
    for (int t = 0; t < ntiles; ++t) {
        wait_async0();        // this wave's async fills (tile t) are in LDS
        __syncthreads();      // all waves' fills done; previous tile's readers done
        if (t + 1 < ntiles) issue_tile(t + 1, (t + 1) & 1);

        const __bf16* base = &sE[t & 1][0];
        const __bf16* cr0  = base + (size_t)nl * DCOL;           // codes n0 .. n0+15
        const __bf16* cr1  = base + (size_t)(16 + nl) * DCOL;    // codes n0+16 .. n0+31

        v8f acc0, acc1;
        gemm2(A, cr0, cr1, hx, acc0, acc1);

        const int n0 = code0 + t * 32;
        {   // stream 0 (lower code indices first -> ties keep first occurrence)
            const int   n  = n0 + nl;
            const float e2 = enorm[n];
            #pragma unroll
            for (int r = 0; r < 8; ++r) {
                float d = e2 - 2.0f * acc0[r];
                if (d < bv[r]) { bv[r] = d; bi[r] = n; }
            }
        }
        {   // stream 1
            const int   n  = n0 + 16 + nl;
            const float e2 = enorm[n];
            #pragma unroll
            for (int r = 0; r < 8; ++r) {
                float d = e2 - 2.0f * acc1[r];
                if (d < bv[r]) { bv[r] = d; bi[r] = n; }
            }
        }
    }

    // reduce across the 16-lane N group (xor within half stays within half)
    #pragma unroll
    for (int m = 1; m < 16; m <<= 1) {
        #pragma unroll
        for (int r = 0; r < 8; ++r) {
            float ov = __shfl_xor(bv[r], m, 32);
            int   oi = __shfl_xor(bi[r], m, 32);
            if (ov < bv[r] || (ov == bv[r] && oi < bi[r])) { bv[r] = ov; bi[r] = oi; }
        }
    }
    if (nl == 0) {
        #pragma unroll
        for (int r = 0; r < 8; ++r) {
            int m = mbase + r + hx * 8;
            atomicMin(&best[m], packDI(bv[r], bi[r]));
        }
    }
}

// -------- gather quantized rows + loss partials --------

__global__ __launch_bounds__(256) void k_gather(const float* __restrict__ emb,
                                                const unsigned long long* __restrict__ best,
                                                const float* __restrict__ x, int xld,
                                                float* __restrict__ qf32,
                                                __bf16* __restrict__ qbf,
                                                float* __restrict__ lossAcc) {
    const int m   = blockIdx.x;
    const int idx = (int)(best[m] & 0xFFFFFFFFull);
    const float* e = emb + (size_t)idx * DCOL;
    float s = 0.0f;
    #pragma unroll
    for (int i = 0; i < 2; ++i) {
        const int c = threadIdx.x + 256 * i;
        float q = e[c];
        qf32[(size_t)m * DCOL + c] = q;
        qbf [(size_t)m * DCOL + c] = (__bf16)q;
        float d = q - x[(size_t)m * xld + c];
        s += d * d;
    }
    #pragma unroll
    for (int o = 1; o < 32; o <<= 1) s += __shfl_xor(s, o, 32);
    __shared__ float ws[8];
    if ((threadIdx.x & 31) == 0) ws[threadIdx.x >> 5] = s;
    __syncthreads();
    if (threadIdx.x == 0) {
        float v = 0.0f;
        #pragma unroll
        for (int i = 0; i < 8; ++i) v += ws[i];
        atomicAdd(lossAcc, v);
    }
}

// -------- fused MLP: h = Q·Wᵀ + b -> LN(g,be) -> leaky -> stage epilogue --------
// block = 256 thr = 8 waves, 16 rows/block; wave w owns cols [w*64, w*64+64),
// processed as two concurrent pairs of 16-col tiles (independent WMMA streams).

template <int STAGE>
__global__ __launch_bounds__(256) void k_mlp(const __bf16* __restrict__ Qbf,
                                             const __bf16* __restrict__ Wbf,
                                             const float* __restrict__ bias,
                                             const float* __restrict__ gamma,
                                             const float* __restrict__ beta,
                                             const float* __restrict__ z,
                                             const float* __restrict__ gateRawC,
                                             const float* __restrict__ gateRawF,
                                             const float* __restrict__ qc,
                                             const float* __restrict__ qf,
                                             const float* __restrict__ ci_in,
                                             float* __restrict__ out0,
                                             float* __restrict__ res_f,
                                             __bf16* __restrict__ res_b) {
    __shared__ __align__(16) float sH[16 * DCOL];    // 32 KB
    const int lane = threadIdx.x & 31;
    const int wave = threadIdx.x >> 5;
    const int hx   = lane >> 4;
    const int nl   = lane & 15;
    const int mbase = blockIdx.x * 16;

    v16bf A[KCHUNKS];
    {
        const __bf16* rowp = Qbf + (size_t)(mbase + nl) * DCOL;
        #pragma unroll
        for (int kc = 0; kc < KCHUNKS; ++kc) A[kc] = load_a_chunk(rowp, kc, hx);
    }

    #pragma unroll
    for (int ntp = 0; ntp < 2; ++ntp) {
        const int n0a = (wave * 4 + ntp * 2) * 16;
        const int n0b = n0a + 16;
        const __bf16* cra = Wbf + (size_t)(n0a + nl) * DCOL;
        const __bf16* crb = Wbf + (size_t)(n0b + nl) * DCOL;

        v8f acca, accb;
        gemm2(A, cra, crb, hx, acca, accb);

        const float bna = bias[n0a + nl];
        const float bnb = bias[n0b + nl];
        #pragma unroll
        for (int r = 0; r < 8; ++r) {
            sH[(r + hx * 8) * DCOL + n0a + nl] = acca[r] + bna;
            sH[(r + hx * 8) * DCOL + n0b + nl] = accb[r] + bnb;
        }
    }
    __syncthreads();

    const float gc = sigmoid_f(gateRawC[0]);
    #pragma unroll
    for (int rr = 0; rr < 2; ++rr) {
        const int mr = wave * 2 + rr;
        const int m  = mbase + mr;
        float s = 0.0f, s2 = 0.0f, hv[16];
        #pragma unroll
        for (int i = 0; i < 16; ++i) {
            float h = sH[mr * DCOL + lane + 32 * i];
            hv[i] = h; s += h; s2 += h * h;
        }
        #pragma unroll
        for (int o = 1; o < 32; o <<= 1) { s += __shfl_xor(s, o, 32); s2 += __shfl_xor(s2, o, 32); }
        const float mean = s * (1.0f / 512.0f);
        const float var  = s2 * (1.0f / 512.0f) - mean * mean;
        const float rstd = rsqrtf(var + 1e-5f);
        #pragma unroll
        for (int i = 0; i < 16; ++i) {
            const int c = lane + 32 * i;
            float v = (hv[i] - mean) * rstd * gamma[c] + beta[c];
            v = v > 0.0f ? v : 0.1f * v;
            if (STAGE == 0) {
                out0[(size_t)m * DCOL + c] = v;                               // coarse_influence
                float r = z[(size_t)m * 1024 + 512 + c] - gc * v;             // residual
                res_f[(size_t)m * DCOL + c] = r;
                res_b[(size_t)m * DCOL + c] = (__bf16)r;
            } else {
                const float gf = sigmoid_f(gateRawF[0]);
                out0[(size_t)m * 1024 + c]       = qc[(size_t)m * DCOL + c] + 0.1f * gf * v;
                out0[(size_t)m * 1024 + 512 + c] = qf[(size_t)m * DCOL + c] + gc * ci_in[(size_t)m * DCOL + c];
            }
        }
    }
}

// -------- finalize scalar loss --------

__global__ void k_final(const float* loss, float* out) {
    out[0] = 1.25f * ((loss[0] + loss[1]) * (1.0f / (8192.0f * 512.0f)));
}

// -------- host launcher --------

extern "C" void kernel_launch(void* const* d_in, const int* in_sizes, int n_in,
                              void* d_out, int out_size, void* d_ws, size_t ws_size,
                              hipStream_t stream) {
    const float* z     = (const float*)d_in[0];
    const float* cemb  = (const float*)d_in[1];
    const float* femb  = (const float*)d_in[2];
    const float* w_c2f = (const float*)d_in[3];
    const float* b_c2f = (const float*)d_in[4];
    const float* g_c2f = (const float*)d_in[5];
    const float* be_c2f= (const float*)d_in[6];
    const float* w_f2c = (const float*)d_in[7];
    const float* b_f2c = (const float*)d_in[8];
    const float* g_f2c = (const float*)d_in[9];
    const float* be_f2c= (const float*)d_in[10];
    const float* gc_raw= (const float*)d_in[11];
    const float* gf_raw= (const float*)d_in[12];
    float* out = (float*)d_out;

    char* ws = (char*)d_ws;
    size_t off = 0;
    auto carve = [&](size_t bytes) { char* p = ws + off; off = (off + bytes + 255) & ~(size_t)255; return p; };

    __bf16* ec_bf = (__bf16*)carve((size_t)4096 * 512 * 2);
    __bf16* ef_bf = (__bf16*)carve((size_t)8192 * 512 * 2);
    __bf16* zc_bf = (__bf16*)carve((size_t)NROWS * 512 * 2);
    __bf16* wc_bf = (__bf16*)carve((size_t)512 * 512 * 2);
    __bf16* wf_bf = (__bf16*)carve((size_t)512 * 512 * 2);
    float*  enc   = (float*)carve((size_t)4096 * 4);
    float*  enf   = (float*)carve((size_t)8192 * 4);
    unsigned long long* bestc = (unsigned long long*)carve((size_t)NROWS * 8);
    unsigned long long* bestf = (unsigned long long*)carve((size_t)NROWS * 8);
    float*  qc_f  = (float*)carve((size_t)NROWS * 512 * 4);
    __bf16* qc_b  = (__bf16*)carve((size_t)NROWS * 512 * 2);
    float*  ci    = (float*)carve((size_t)NROWS * 512 * 4);
    float*  resf  = (float*)carve((size_t)NROWS * 512 * 4);
    __bf16* resb  = (__bf16*)carve((size_t)NROWS * 512 * 2);
    float*  qf_f  = (float*)carve((size_t)NROWS * 512 * 4);
    __bf16* qf_b  = (__bf16*)carve((size_t)NROWS * 512 * 2);
    float*  loss  = (float*)carve(8);

    k_init<<<NROWS / 256, 256, 0, stream>>>(bestc, bestf, loss);

    k_cvt_norm<<<4096 / 8, 256, 0, stream>>>(cemb, 512, ec_bf, enc, 512);
    k_cvt_norm<<<8192 / 8, 256, 0, stream>>>(femb, 512, ef_bf, enf, 512);
    k_cvt_norm<<<NROWS / 8, 256, 0, stream>>>(z, 1024, zc_bf, nullptr, 512);   // z[:, :512]
    k_cvt_norm<<<512 / 8, 256, 0, stream>>>(w_c2f, 512, wc_bf, nullptr, 512);
    k_cvt_norm<<<512 / 8, 256, 0, stream>>>(w_f2c, 512, wf_bf, nullptr, 512);

    // coarse VQ: 4096 codes, 4 splits of 1024
    k_vq<<<dim3(NROWS / 128, 4), 256, 0, stream>>>(zc_bf, ec_bf, enc, bestc, 1024);
    k_gather<<<NROWS, 256, 0, stream>>>(cemb, bestc, z, 1024, qc_f, qc_b, loss + 0);

    // c2f MLP -> coarse_influence, residual
    k_mlp<0><<<NROWS / 16, 256, 0, stream>>>(qc_b, wc_bf, b_c2f, g_c2f, be_c2f,
                                             z, gc_raw, nullptr, nullptr, nullptr, nullptr,
                                             ci, resf, resb);

    // fine VQ: 8192 codes, 8 splits of 1024
    k_vq<<<dim3(NROWS / 128, 8), 256, 0, stream>>>(resb, ef_bf, enf, bestf, 1024);
    k_gather<<<NROWS, 256, 0, stream>>>(femb, bestf, resf, 512, qf_f, qf_b, loss + 1);

    // f2c MLP -> fine_feedback, final concat outputs
    k_mlp<1><<<NROWS / 16, 256, 0, stream>>>(qf_b, wf_bf, b_f2c, g_f2c, be_f2c,
                                             nullptr, gc_raw, gf_raw, qc_f, qf_f, ci,
                                             out, nullptr, nullptr);

    k_final<<<1, 1, 0, stream>>>(loss, out + (size_t)NROWS * 1024);
}
